// GAT_46600395162312
// MI455X (gfx1250) — matrix operand order repaired and checked by reading, not compiled, not run on hardware
//
#include <hip/hip_runtime.h>

typedef __attribute__((ext_vector_type(16))) _Float16 v16h;
typedef __attribute__((ext_vector_type(8)))  float    v8f;
typedef __attribute__((ext_vector_type(4)))  unsigned u32x4;
typedef __attribute__((ext_vector_type(8)))  unsigned u32x8;

#define GN      12288
#define GIN     256
#define GOUT    128
#define NWORDS  (GN / 32)      // 384 packed words per row
#define BJ      128            // columns staged per LDS phase (4 WMMA K-tiles)
#define NEGS    0.01f

__device__ __forceinline__ float leaky(float x) { return x > 0.f ? x : NEGS * x; }

// ---------------------------------------------------------------------------
// K1: new_features = X @ W^T + b   via v_wmma_f32_16x16x32_f16
// ---------------------------------------------------------------------------
__global__ void __launch_bounds__(256) proj_kernel(
    const float* __restrict__ X, const float* __restrict__ W,
    const float* __restrict__ bias, float* __restrict__ nf,
    _Float16* __restrict__ nfh) {
  const int r0   = blockIdx.x * 16;
  const int wv   = threadIdx.x >> 5;
  const int lane = threadIdx.x & 31;
  const int n0   = wv * 16;
  const int hi   = lane >> 4;
  const int ml   = lane & 15;

  v8f acc = {};
  for (int k0 = 0; k0 < GIN; k0 += 32) {
    v16h a, b;
    const float* xrow = X + (size_t)(r0 + ml) * GIN;
#pragma unroll
    for (int i = 0; i < 8; ++i) {
      int k = k0 + ((i >= 4) ? 16 : 0) + (hi ? 8 : 0) + (i & 3) * 2;
      a[2 * i]     = (_Float16)xrow[k];
      a[2 * i + 1] = (_Float16)xrow[k + 1];
    }
#pragma unroll
    for (int i = 0; i < 8; ++i) {
      int k = k0 + ((i >= 4) ? 16 : 0) + (hi ? 8 : 0) + (i & 3) * 2;
      b[2 * i]     = (_Float16)W[(size_t)(n0 + ml) * GIN + k];
      b[2 * i + 1] = (_Float16)W[(size_t)(n0 + ml) * GIN + k + 1];
    }
    acc = __builtin_amdgcn_wmma_f32_16x16x32_f16(false, a, false, b,
                                                 (short)0, acc, false, false);
  }
  const int nn = n0 + ml;
  const float bv = bias[nn];
#pragma unroll
  for (int v = 0; v < 8; ++v) {
    int m = v + (hi ? 8 : 0);
    float val = acc[v] + bv;
    nf[(size_t)(r0 + m) * GOUT + nn]  = val;
    nfh[(size_t)(r0 + m) * GOUT + nn] = (_Float16)val;
  }
}

// ---------------------------------------------------------------------------
// K1b: f1 = nf @ phi[:128], f2 = nf @ phi[128:]
// ---------------------------------------------------------------------------
__global__ void f1f2_kernel(const float* __restrict__ nf,
                            const float* __restrict__ phi,
                            float* __restrict__ f1, float* __restrict__ f2) {
  int row = blockIdx.x * blockDim.x + threadIdx.x;
  if (row >= GN) return;
  const float* p = nf + (size_t)row * GOUT;
  float a = 0.f, b = 0.f;
#pragma unroll 4
  for (int c = 0; c < GOUT; ++c) {
    float v = p[c];
    a += v * phi[c];
    b += v * phi[GOUT + c];
  }
  f1[row] = a;
  f2[row] = b;
}

// ---------------------------------------------------------------------------
// K2: stream adjacency ONCE per row (non-temporal; no reuse): online softmax
// stats over edges + self loop, pack edge bits with wave32 ballot.
// ---------------------------------------------------------------------------
__global__ void __launch_bounds__(256) stats_kernel(
    const int* __restrict__ adj, const float* __restrict__ f1,
    const float* __restrict__ f2, float* __restrict__ mrow,
    float* __restrict__ invZ, unsigned* __restrict__ packed) {
  const int row = blockIdx.x;
  const int tid = threadIdx.x;
  const float f1i = f1[row];
  const int* arow = adj + (size_t)row * GN;

  float mt = -3.0e38f, zt = 0.f;
  for (int cb = 0; cb < GN; cb += 256) {
    __builtin_prefetch(arow + cb + 2048, 0, 0);   // global_prefetch_b8
    int col = cb + tid;
    int av = __builtin_nontemporal_load(arow + col);
    bool edge = (av > 0) || (col == row);         // adj + eye > 0
    unsigned msk = __builtin_amdgcn_ballot_w32(edge);
    if ((tid & 31) == 0) packed[(size_t)row * NWORDS + ((cb + tid) >> 5)] = msk;
    if (edge) {
      float s = leaky(f1i + f2[col]);
      if (s > mt) { zt = zt * __expf(mt - s) + 1.f; mt = s; }
      else        { zt += __expf(s - mt); }
    }
  }
  __shared__ float smx[256], ssm[256];
  smx[tid] = mt; ssm[tid] = zt;
  __syncthreads();
  for (int off = 128; off > 0; off >>= 1) {
    if (tid < off) {
      float m1 = smx[tid],       z1 = ssm[tid];
      float m2 = smx[tid + off], z2 = ssm[tid + off];
      float mm = fmaxf(m1, m2);
      smx[tid] = mm;
      ssm[tid] = z1 * __expf(m1 - mm) + z2 * __expf(m2 - mm);
    }
    __syncthreads();
  }
  if (tid == 0) { mrow[row] = smx[0]; invZ[row] = 1.f / ssm[0]; }
}

// ---------------------------------------------------------------------------
// K3: out = softmax(masked sim) @ nf, fused. Per 16-row block, stage 128
// columns at a time: TDM (tensor_load_to_lds) DMAs the 128x128-f16 V tile
// into LDS while all waves compute the attention-weight tile P; then each of
// the 8 waves runs 4 back-to-back WMMAs per barrier period.
// ---------------------------------------------------------------------------
__global__ void __launch_bounds__(256) attnv_kernel(
    const unsigned* __restrict__ packed, const _Float16* __restrict__ nfh,
    const float* __restrict__ f1, const float* __restrict__ f2,
    const float* __restrict__ mrow, const float* __restrict__ invZ,
    float* __restrict__ out) {
  const int r0   = blockIdx.x * 16;
  const int tid  = threadIdx.x;
  const int wv   = tid >> 5;
  const int lane = tid & 31;
  const int hi   = lane >> 4;
  const int ml   = lane & 15;
  const int n0   = wv * 16;

  __shared__ __align__(16) _Float16 Vt[BJ][GOUT];  // 128x128 f16 = 32KB (TDM dest)
  __shared__ __align__(16) _Float16 Pp[16][BJ];    // 16x128 f16  = 4KB
  __shared__ unsigned swrd[16 * (BJ / 32)];        // 64 mask words
  __shared__ float sf1[16], sm[16], siz[16];

  if (tid < 16) {
    sf1[tid] = f1[r0 + tid];
    sm[tid]  = mrow[r0 + tid];
    siz[tid] = invZ[r0 + tid];
  }

  // Tensor DMA descriptor (CDNA5 D#), constant parts. 2D tile:
  //   data_size=4B, tile_dim0=64 dwords (=128 f16), tile_dim1=128 rows,
  //   tensor_dim0=64, tensor_dim1=GN, tensor_dim0_stride=64 dwords.
  const unsigned ldsa = (unsigned)(size_t)(&Vt[0][0]);
  u32x8 g1;
  g1[0] = 0x00020000u;            // workgroup_mask=0, data_size=2 (4B)
  g1[1] = 64u << 16;              // tensor_dim0[15:0]=64 at bits 79:64... (63:48)
  g1[2] = (unsigned)GN << 16;     // tensor_dim1[15:0] at bits 95:80 (d2[31:16])
  g1[3] = 64u << 16;              // tile_dim0=64 at bits 127:112
  g1[4] = (unsigned)BJ;           // tile_dim1=128 rows
  g1[5] = 64u;                    // tensor_dim0_stride=64 dwords
  g1[6] = 0u;
  g1[7] = 0u;
  __syncthreads();

  v8f acc = {};
  for (int j0 = 0; j0 < GN; j0 += BJ) {
    // ---- phase A: kick off TDM for the V tile; fetch mask words ----
    if (wv == 0) {
      unsigned long long ga =
          (unsigned long long)(size_t)nfh + (unsigned long long)j0 * (GOUT * 2);
      u32x4 g0;
      g0[0] = 1u;                                       // count=1 (valid D#)
      g0[1] = ldsa;                                     // lds_addr
      g0[2] = (unsigned)(ga & 0xffffffffu);             // global_addr[31:0]
      g0[3] = (unsigned)((ga >> 32) & 0x01ffffffu) | (2u << 30); // addr[56:32]|type=2
      asm volatile("tensor_load_to_lds %0, %1" :: "s"(g0), "s"(g1) : "memory");
    }
    if (tid < 64)
      swrd[tid] = packed[(size_t)(r0 + (tid >> 2)) * NWORDS + (j0 >> 5) + (tid & 3)];
    __syncthreads();

    // ---- phase B: compute P[16][128] while the DMA is in flight ----
#pragma unroll
    for (int e = tid; e < 16 * BJ; e += 256) {
      int il = e >> 7, jl = e & (BJ - 1);
      float p = 0.f;
      if (swrd[il * 4 + (jl >> 5)] & (1u << (jl & 31))) {
        float s = leaky(sf1[il] + f2[j0 + jl]);
        p = __expf(s - sm[il]) * siz[il];
      }
      Pp[il][jl] = (_Float16)p;
    }
    if (wv == 0) __builtin_amdgcn_s_wait_tensorcnt(0);  // V tile landed in LDS
    __syncthreads();

    // ---- phase C: 4 WMMA K-subtiles per wave ----
#pragma unroll
    for (int kk = 0; kk < 4; ++kk) {
      const int ko = kk * 32;
      v16h a, b;
#pragma unroll
      for (int i = 0; i < 8; ++i) {
        int k = ko + ((i >= 4) ? 16 : 0) + (hi ? 8 : 0) + (i & 3) * 2;
        a[2 * i]     = Pp[ml][k];
        a[2 * i + 1] = Pp[ml][k + 1];
      }
#pragma unroll
      for (int i = 0; i < 8; ++i) {
        int k = ko + ((i >= 4) ? 16 : 0) + (hi ? 8 : 0) + (i & 3) * 2;
        b[2 * i]     = Vt[k][n0 + ml];
        b[2 * i + 1] = Vt[k + 1][n0 + ml];
      }
      acc = __builtin_amdgcn_wmma_f32_16x16x32_f16(false, a, false, b,
                                                   (short)0, acc, false, false);
    }
    __syncthreads();
  }

  const int nn = n0 + ml;
#pragma unroll
  for (int v = 0; v < 8; ++v) {
    int m = v + (hi ? 8 : 0);
    out[(size_t)(r0 + m) * GOUT + nn] = leaky(acc[v]);
  }
}

// ---------------------------------------------------------------------------
extern "C" void kernel_launch(void* const* d_in, const int* in_sizes, int n_in,
                              void* d_out, int out_size, void* d_ws, size_t ws_size,
                              hipStream_t stream) {
  const int*   adj  = (const int*)d_in[0];    // [N,N] int32
  const float* X    = (const float*)d_in[1];  // [N,256]
  const float* W    = (const float*)d_in[2];  // [128,256]
  const float* bias = (const float*)d_in[3];  // [128]
  const float* phi  = (const float*)d_in[4];  // [256]
  float* out = (float*)d_out;                 // [N,128]

  char* w = (char*)d_ws;
  float*    nf     = (float*)w;      w += (size_t)GN * GOUT * sizeof(float);
  _Float16* nfh    = (_Float16*)w;   w += (size_t)GN * GOUT * sizeof(_Float16);
  float*    f1     = (float*)w;      w += (size_t)GN * sizeof(float);
  float*    f2     = (float*)w;      w += (size_t)GN * sizeof(float);
  float*    mr     = (float*)w;      w += (size_t)GN * sizeof(float);
  float*    iz     = (float*)w;      w += (size_t)GN * sizeof(float);
  unsigned* packed = (unsigned*)w;   // GN * NWORDS u32 = 18.9 MB

  proj_kernel<<<GN / 16, 256, 0, stream>>>(X, W, bias, nf, nfh);
  f1f2_kernel<<<GN / 256, 256, 0, stream>>>(nf, phi, f1, f2);
  stats_kernel<<<GN, 256, 0, stream>>>(adj, f1, f2, mr, iz, packed);
  attnv_kernel<<<GN / 16, 256, 0, stream>>>(packed, nfh, f1, f2, mr, iz, out);
}
